// AttentionDecoderLayer_62337155334503
// MI455X (gfx1250) — matrix operand order repaired and checked by reading, not compiled, not run on hardware
//
#include <hip/hip_runtime.h>

typedef __attribute__((ext_vector_type(16))) _Float16 v16h;
typedef __attribute__((ext_vector_type(8)))  _Float16 v8h;
typedef __attribute__((ext_vector_type(8)))  float    v8f;

#define B_   2
#define L_   2048
#define DM   1024
#define H_   16
#define HD_  64
#define DFF_ 4096
#define NROWS (B_*L_)   /* 4096 */
#define KCH  32         /* GEMM k-chunk = one WMMA K-step */

// ---------------------------------------------------------------------------
// WMMA helpers (gfx1250 wave32; A/B use this file's contiguous k order, not the ISA table order)
// A fragment (16x32 f16): lane l holds row m=l&15; elem e is
// K=16*hi+e, hi=l>>4.
// B fragment (32x16 f16): lane l holds col n=l&15; elem e is K=16*hi+e.
// C/D (16x16 f32): lane l col n=l&15; VGPR v is row m=v+8*hi.
// ---------------------------------------------------------------------------
__device__ __forceinline__ v16h make_a_frag(const _Float16* row, int hi) {
  v8h lo8 = *(const v8h*)(row + hi * 16);
  v8h hi8 = *(const v8h*)(row + hi * 16 + 8);
  return __builtin_shufflevector(lo8, hi8, 0,1,2,3,4,5,6,7,8,9,10,11,12,13,14,15);
}

__device__ __forceinline__ v16h load_b_frag(const _Float16* p) {
  v8h a = *(const v8h*)p;
  v8h b = *(const v8h*)(p + 8);
  return __builtin_shufflevector(a, b, 0,1,2,3,4,5,6,7,8,9,10,11,12,13,14,15);
}

__device__ __forceinline__ v8f wmma_f16(v16h a, v16h b, v8f c) {
  return __builtin_amdgcn_wmma_f32_16x16x32_f16(false, a, false, b, (short)0, c,
                                                false, false);
}

__device__ __forceinline__ v8f zero8() {
  v8f z = {0.f, 0.f, 0.f, 0.f, 0.f, 0.f, 0.f, 0.f};
  return z;
}

// ---------------------------------------------------------------------------
// CDNA5 async copy: global -> LDS, 16B per lane, tracked by ASYNCcnt.
// ---------------------------------------------------------------------------
__device__ __forceinline__ unsigned lds_off(const void* p) {
  return (unsigned)(unsigned long long)(uintptr_t)p;  // low 32b = LDS offset
}

__device__ __forceinline__ void async_b128(unsigned dst, const void* src) {
  asm volatile("global_load_async_to_lds_b128 %0, %1, off"
               :: "v"(dst), "v"(src) : "memory");
}

#define WAIT_ASYNC_LE2() asm volatile("s_wait_asynccnt 0x2" ::: "memory")

// ---------------------------------------------------------------------------
// Elementwise cast f32 -> f16
// ---------------------------------------------------------------------------
__global__ void cast_kernel(const float* __restrict__ in,
                            _Float16* __restrict__ out, int n) {
  int idx = blockIdx.x * blockDim.x + threadIdx.x;
  if (idx < n) out[idx] = (_Float16)in[idx];
}

// ---------------------------------------------------------------------------
// Weight transpose + cast: W (K x N, f32) -> Wt (N x K, f16)
// ---------------------------------------------------------------------------
__global__ void wcast_kernel(const float* __restrict__ W,
                             _Float16* __restrict__ Wt, int K, int N) {
  size_t idx = (size_t)blockIdx.x * blockDim.x + threadIdx.x;
  if (idx >= (size_t)K * N) return;
  int k = (int)(idx / N);
  int n = (int)(idx % N);
  Wt[(size_t)n * K + k] = (_Float16)W[idx];
}

// ---------------------------------------------------------------------------
// LayerNorm over last dim (DM=1024), one block per row, f16 output
// ---------------------------------------------------------------------------
__global__ void ln_kernel(const float* __restrict__ x,
                          const float* __restrict__ g,
                          const float* __restrict__ b,
                          _Float16* __restrict__ outh) {
  __shared__ float s1[256], s2[256];
  int r = blockIdx.x, t = threadIdx.x;
  const float* row = x + (size_t)r * DM;
  float sum = 0.f, sq = 0.f;
  for (int i = t; i < DM; i += 256) {
    float v = row[i];
    sum += v; sq += v * v;
  }
  s1[t] = sum; s2[t] = sq;
  __syncthreads();
  for (int s = 128; s > 0; s >>= 1) {
    if (t < s) { s1[t] += s1[t + s]; s2[t] += s2[t + s]; }
    __syncthreads();
  }
  float mu   = s1[0] * (1.f / DM);
  float var  = s2[0] * (1.f / DM) - mu * mu;
  float rinv = rsqrtf(var + 1e-5f);
  for (int i = t; i < DM; i += 256) {
    float v = (row[i] - mu) * rinv * g[i] + b[i];
    outh[(size_t)r * DM + i] = (_Float16)v;
  }
}

// ---------------------------------------------------------------------------
// Pipelined WMMA GEMM: C(MxN) = A(MxK,f16) @ Wt^T (Wt is NxK f16) + bias.
// Block = 128 thr = 4 waves; block tile 128(M) x 64(N); wave tile 32x64.
// B tile (64x32 = 4KB) staged in LDS by async copy, double-buffered,
// shared by all 4 waves. A fragments ping-ponged in registers.
// __launch_bounds__(128, 1): let RA keep all accumulators resident.
// ---------------------------------------------------------------------------
__global__ void __launch_bounds__(128, 1)
gemm_wmma(const _Float16* __restrict__ A,
          const _Float16* __restrict__ Wt,
          const float* __restrict__ bias,
          const float* __restrict__ resid,
          float* __restrict__ outf,
          _Float16* __restrict__ outh,
          int M, int N, int K, int relu) {
  __shared__ alignas(128) _Float16 Blds[2][64][KCH];
  int tid  = threadIdx.x;
  int wave = tid >> 5, lane = tid & 31;
  int lo = lane & 15, hi = lane >> 4;
  int m0 = blockIdx.y * 128 + wave * 32;
  int n0 = blockIdx.x * 64;

  // async staging map: 256 x 16B segments; thread t owns segments t, t+128
  int sa = tid, sb = tid + 128;
  int ca = sa >> 2, ka = (sa & 3) * 8;
  int cb = sb >> 2, kb = (sb & 3) * 8;
  const _Float16* gBa = Wt + (size_t)(n0 + ca) * K + ka;
  const _Float16* gBb = Wt + (size_t)(n0 + cb) * K + kb;
  unsigned da[2], db[2];
  da[0] = lds_off(&Blds[0][ca][ka]);
  da[1] = lds_off(&Blds[1][ca][ka]);
  db[0] = lds_off(&Blds[0][cb][kb]);
  db[1] = lds_off(&Blds[1][cb][kb]);

  v8f acc[2][4];
#pragma unroll
  for (int ms = 0; ms < 2; ++ms)
#pragma unroll
    for (int nt = 0; nt < 4; ++nt) acc[ms][nt] = zero8();

  const _Float16* arow0 = A + (size_t)(m0 + lo) * K;
  const _Float16* arow1 = A + (size_t)(m0 + 16 + lo) * K;

  int nch = K / KCH;
  // prologue: chunks 0 and 1 in flight
  async_b128(da[0], gBa);        async_b128(db[0], gBb);
  async_b128(da[1], gBa + KCH);  async_b128(db[1], gBb + KCH);
  v16h a0 = make_a_frag(arow0, hi);
  v16h a1 = make_a_frag(arow1, hi);

  for (int i = 0; i < nch; ++i) {
    WAIT_ASYNC_LE2();          // chunk i complete (chunk i+1 still in flight)
    __syncthreads();
    int buf = i & 1;
    v16h bb0 = load_b_frag(&Blds[buf][0 * 16 + lo][hi * 16]);
    v16h bb1 = load_b_frag(&Blds[buf][1 * 16 + lo][hi * 16]);
    v16h bb2 = load_b_frag(&Blds[buf][2 * 16 + lo][hi * 16]);
    v16h bb3 = load_b_frag(&Blds[buf][3 * 16 + lo][hi * 16]);
    // prefetch next chunk's A fragments (in flight during the 8 WMMAs)
    int kn = (i + 1 < nch) ? (i + 1) * KCH : 0;
    v16h na0 = make_a_frag(arow0 + kn, hi);
    v16h na1 = make_a_frag(arow1 + kn, hi);

    acc[0][0] = wmma_f16(a0, bb0, acc[0][0]);
    acc[1][0] = wmma_f16(a1, bb0, acc[1][0]);
    acc[0][1] = wmma_f16(a0, bb1, acc[0][1]);
    acc[1][1] = wmma_f16(a1, bb1, acc[1][1]);
    acc[0][2] = wmma_f16(a0, bb2, acc[0][2]);
    acc[1][2] = wmma_f16(a1, bb2, acc[1][2]);
    acc[0][3] = wmma_f16(a0, bb3, acc[0][3]);
    acc[1][3] = wmma_f16(a1, bb3, acc[1][3]);

    __syncthreads();           // all waves done reading buf
    // keep the 2-in-flight invariant: issue chunk i+2 into buf (wrap = dummy)
    int kp = (i + 2 < nch) ? (i + 2) * KCH : 0;
    async_b128(da[buf], gBa + kp);
    async_b128(db[buf], gBb + kp);
    a0 = na0; a1 = na1;
  }

#pragma unroll
  for (int ms = 0; ms < 2; ++ms)
#pragma unroll
    for (int nt = 0; nt < 4; ++nt) {
      int col = n0 + nt * 16 + lo;
      float bv = bias ? bias[col] : 0.f;
#pragma unroll
      for (int v = 0; v < 8; ++v) {
        int row = m0 + ms * 16 + v + 8 * hi;
        float val = acc[ms][nt][v] + bv;
        if (relu) val = fmaxf(val, 0.f);
        size_t idx = (size_t)row * N + col;
        if (resid) val += resid[idx];
        if (outf) outf[idx] = val;
        if (outh) outh[idx] = (_Float16)val;
      }
    }
}

// ---------------------------------------------------------------------------
// RoPE (first 8 heads of 16) + 1/sqrt(HD) scaling of all Q, in place on f16.
// ---------------------------------------------------------------------------
__global__ void rope_kernel(_Float16* __restrict__ Qh, _Float16* __restrict__ Kh) {
  int idx = blockIdx.x * blockDim.x + threadIdx.x;
  if (idx >= NROWS * H_ * 32) return;
  int i = idx & 31;
  int h = (idx >> 5) & 15;
  int r = idx >> 9;
  int pos = r & (L_ - 1);
  float freq = __expf(-(float)i * (9.210340371976184f / 32.0f));
  float ang = (float)pos * freq;
  float c = __cosf(ang), s = __sinf(ang);
  size_t base = (size_t)r * DM + h * HD_ + 2 * i;
  float q1 = (float)Qh[base], q2 = (float)Qh[base + 1];
  float qe, qo;
  if (h < 8) { qe = q1 * c - q2 * s; qo = q1 * s + q2 * c; }
  else       { qe = q1;              qo = q2; }
  Qh[base]     = (_Float16)(qe * 0.125f);
  Qh[base + 1] = (_Float16)(qo * 0.125f);
  if (h < 8) {
    float k1 = (float)Kh[base], k2 = (float)Kh[base + 1];
    Kh[base]     = (_Float16)(k1 * c - k2 * s);
    Kh[base + 1] = (_Float16)(k1 * s + k2 * c);
  }
}

// ---------------------------------------------------------------------------
// V transpose: Vh (B*L x DM) -> Vth (B x DM x L)
// ---------------------------------------------------------------------------
__global__ void vtrans_kernel(const _Float16* __restrict__ Vh,
                              _Float16* __restrict__ Vth) {
  int idx = blockIdx.x * blockDim.x + threadIdx.x;
  if (idx >= NROWS * DM) return;
  int r = idx / DM, c = idx % DM;
  int b = r / L_, l = r % L_;
  Vth[((size_t)(b * DM + c)) * L_ + l] = Vh[idx];
}

// ---------------------------------------------------------------------------
// Flash attention, 32 query rows per wave: 8 score WMMAs, online softmax,
// P re-layout via per-wave LDS slice (same-wave DS, in-order), 8 PV WMMAs.
// ---------------------------------------------------------------------------
__global__ void __launch_bounds__(128, 1)
attn_kernel(const _Float16* __restrict__ Qh,
            const _Float16* __restrict__ Kh,
            const _Float16* __restrict__ Vth,
            _Float16* __restrict__ Oh) {
  __shared__ alignas(64) _Float16 Plds[4][2][16][32];
  int wave = threadIdx.x >> 5;
  int lane = threadIdx.x & 31;
  int lo = lane & 15, hi = lane >> 4;
  int b = blockIdx.y >> 4, h = blockIdx.y & 15;
  int q0 = blockIdx.x * 128 + wave * 32;

  v16h qa[2][2];
#pragma unroll
  for (int ms = 0; ms < 2; ++ms) {
    const _Float16* qrow =
        Qh + ((size_t)(b * L_ + q0 + ms * 16 + lo)) * DM + h * HD_;
    qa[ms][0] = make_a_frag(qrow, hi);
    qa[ms][1] = make_a_frag(qrow + 32, hi);
  }

  float m_run[2][8], l_run[2][8];
  v8f o[2][4];
#pragma unroll
  for (int ms = 0; ms < 2; ++ms) {
#pragma unroll
    for (int v = 0; v < 8; ++v) { m_run[ms][v] = -1e30f; l_run[ms][v] = 0.f; }
#pragma unroll
    for (int dt = 0; dt < 4; ++dt) o[ms][dt] = zero8();
  }

  for (int kt = 0; kt < L_ / 32; ++kt) {
    const _Float16* kbase = Kh + ((size_t)(b * L_ + kt * 32)) * DM + h * HD_;
    const _Float16* k0row = kbase + (size_t)lo * DM;
    const _Float16* k1row = kbase + (size_t)(16 + lo) * DM;
    v16h kb00 = load_b_frag(k0row + hi * 16);
    v16h kb01 = load_b_frag(k0row + 32 + hi * 16);
    v16h kb10 = load_b_frag(k1row + hi * 16);
    v16h kb11 = load_b_frag(k1row + 32 + hi * 16);

    v8f s[2][2];
#pragma unroll
    for (int ms = 0; ms < 2; ++ms) {
      s[ms][0] = zero8(); s[ms][1] = zero8();
      s[ms][0] = wmma_f16(qa[ms][0], kb00, s[ms][0]);
      s[ms][0] = wmma_f16(qa[ms][1], kb01, s[ms][0]);
      s[ms][1] = wmma_f16(qa[ms][0], kb10, s[ms][1]);
      s[ms][1] = wmma_f16(qa[ms][1], kb11, s[ms][1]);
    }

    // V fragments issued here: in flight during softmax VALU work
    v16h vb[4];
#pragma unroll
    for (int dt = 0; dt < 4; ++dt)
      vb[dt] = load_b_frag(Vth +
          ((size_t)(b * DM + h * HD_ + dt * 16 + lo)) * L_ + kt * 32 + hi * 16);

    v16h pa[2];
#pragma unroll
    for (int ms = 0; ms < 2; ++ms) {
      float mt[8], al[8], rs[8];
#pragma unroll
      for (int v = 0; v < 8; ++v) mt[v] = fmaxf(s[ms][0][v], s[ms][1][v]);
#pragma unroll
      for (int m = 1; m < 16; m <<= 1)
#pragma unroll
        for (int v = 0; v < 8; ++v)
          mt[v] = fmaxf(mt[v], __shfl_xor(mt[v], m, 32));
#pragma unroll
      for (int v = 0; v < 8; ++v) {
        float mn = fmaxf(m_run[ms][v], mt[v]);
        al[v] = __expf(m_run[ms][v] - mn);
        m_run[ms][v] = mn;
        float p0 = __expf(s[ms][0][v] - mn);
        float p1 = __expf(s[ms][1][v] - mn);
        s[ms][0][v] = p0; s[ms][1][v] = p1;
        rs[v] = p0 + p1;
      }
#pragma unroll
      for (int m = 1; m < 16; m <<= 1)
#pragma unroll
        for (int v = 0; v < 8; ++v) rs[v] += __shfl_xor(rs[v], m, 32);
#pragma unroll
      for (int v = 0; v < 8; ++v)
        l_run[ms][v] = l_run[ms][v] * al[v] + rs[v];
#pragma unroll
      for (int dt = 0; dt < 4; ++dt)
#pragma unroll
        for (int v = 0; v < 8; ++v) o[ms][dt][v] *= al[v];
      // C-layout -> LDS [m][k] -> A fragment (same-wave DS, in-order)
#pragma unroll
      for (int v = 0; v < 8; ++v) {
        Plds[wave][ms][v + 8 * hi][lo]      = (_Float16)s[ms][0][v];
        Plds[wave][ms][v + 8 * hi][16 + lo] = (_Float16)s[ms][1][v];
      }
      pa[ms] = make_a_frag(&Plds[wave][ms][lo][0], hi);
    }

#pragma unroll
    for (int ms = 0; ms < 2; ++ms)
#pragma unroll
      for (int dt = 0; dt < 4; ++dt)
        o[ms][dt] = wmma_f16(pa[ms], vb[dt], o[ms][dt]);
  }

#pragma unroll
  for (int ms = 0; ms < 2; ++ms)
#pragma unroll
    for (int dt = 0; dt < 4; ++dt) {
      int col = h * HD_ + dt * 16 + lo;
#pragma unroll
      for (int v = 0; v < 8; ++v) {
        int row = q0 + ms * 16 + v + 8 * hi;
        float val = o[ms][dt][v] / l_run[ms][v];
        Oh[((size_t)(b * L_ + row)) * DM + col] = (_Float16)val;
      }
    }
}

// ---------------------------------------------------------------------------
// Host-side orchestration
// ---------------------------------------------------------------------------
extern "C" void kernel_launch(void* const* d_in, const int* in_sizes, int n_in,
                              void* d_out, int out_size, void* d_ws, size_t ws_size,
                              hipStream_t stream) {
  const float* x    = (const float*)d_in[0];
  const float* Wq   = (const float*)d_in[1];
  const float* bq   = (const float*)d_in[2];
  const float* Wk   = (const float*)d_in[3];
  const float* bk   = (const float*)d_in[4];
  const float* Wv   = (const float*)d_in[5];
  const float* bv   = (const float*)d_in[6];
  const float* Wo   = (const float*)d_in[7];
  const float* bo   = (const float*)d_in[8];
  const float* ln1g = (const float*)d_in[9];
  const float* ln1b = (const float*)d_in[10];
  const float* ln2g = (const float*)d_in[11];
  const float* ln2b = (const float*)d_in[12];
  const float* W1   = (const float*)d_in[13];
  const float* b1   = (const float*)d_in[14];
  const float* W2   = (const float*)d_in[15];
  const float* b2   = (const float*)d_in[16];
  float* out = (float*)d_out;

  char* ws = (char*)d_ws;
  size_t off = 0;
  auto alloc = [&](size_t bytes) -> char* {
    char* p = ws + off;
    off += (bytes + 255) & ~(size_t)255;
    return p;
  };

  const size_t RD = (size_t)NROWS * DM;  // 4M elements
  _Float16* xh   = (_Float16*)alloc(RD * 2);
  _Float16* xnh  = (_Float16*)alloc(RD * 2);
  _Float16* Wqt  = (_Float16*)alloc((size_t)DM * DM * 2);
  _Float16* Wkt  = (_Float16*)alloc((size_t)DM * DM * 2);
  _Float16* Wvt  = (_Float16*)alloc((size_t)DM * DM * 2);
  _Float16* Wot  = (_Float16*)alloc((size_t)DM * DM * 2);
  _Float16* W1t  = (_Float16*)alloc((size_t)DM * DFF_ * 2);
  _Float16* W2t  = (_Float16*)alloc((size_t)DFF_ * DM * 2);
  _Float16* Qh   = (_Float16*)alloc(RD * 2);
  _Float16* Kh   = (_Float16*)alloc(RD * 2);
  _Float16* Vh   = (_Float16*)alloc(RD * 2);
  _Float16* Vth  = (_Float16*)alloc(RD * 2);
  _Float16* Ohm  = (_Float16*)alloc(RD * 2);
  float*    x2   = (float*)   alloc(RD * 4);
  _Float16* xn2h = (_Float16*)alloc(RD * 2);
  _Float16* H1h  = (_Float16*)alloc((size_t)NROWS * DFF_ * 2);

  cast_kernel<<<(int)((RD + 255) / 256), 256, 0, stream>>>(x, xh, (int)RD);
  ln_kernel<<<NROWS, 256, 0, stream>>>(x, ln1g, ln1b, xnh);
  wcast_kernel<<<(DM * DM + 255) / 256, 256, 0, stream>>>(Wq, Wqt, DM, DM);
  wcast_kernel<<<(DM * DM + 255) / 256, 256, 0, stream>>>(Wk, Wkt, DM, DM);
  wcast_kernel<<<(DM * DM + 255) / 256, 256, 0, stream>>>(Wv, Wvt, DM, DM);
  wcast_kernel<<<(DM * DM + 255) / 256, 256, 0, stream>>>(Wo, Wot, DM, DM);
  wcast_kernel<<<(DM * DFF_ + 255) / 256, 256, 0, stream>>>(W1, W1t, DM, DFF_);
  wcast_kernel<<<(DFF_ * DM + 255) / 256, 256, 0, stream>>>(W2, W2t, DFF_, DM);

  dim3 gblk(128);
  // Q = ln1(x) @ Wq + bq
  gemm_wmma<<<dim3(DM / 64, NROWS / 128), gblk, 0, stream>>>(
      xnh, Wqt, bq, nullptr, nullptr, Qh, NROWS, DM, DM, 0);
  // K = x @ Wk + bk
  gemm_wmma<<<dim3(DM / 64, NROWS / 128), gblk, 0, stream>>>(
      xh, Wkt, bk, nullptr, nullptr, Kh, NROWS, DM, DM, 0);
  // V = x @ Wv + bv
  gemm_wmma<<<dim3(DM / 64, NROWS / 128), gblk, 0, stream>>>(
      xh, Wvt, bv, nullptr, nullptr, Vh, NROWS, DM, DM, 0);
  // RoPE (heads 0..7) + 0.125 scale on Q
  rope_kernel<<<(NROWS * H_ * 32 + 255) / 256, 256, 0, stream>>>(Qh, Kh);
  // V transpose for PV B-fragments
  vtrans_kernel<<<(int)((RD + 255) / 256), 256, 0, stream>>>(Vh, Vth);
  // flash attention
  attn_kernel<<<dim3(L_ / 128, B_ * H_), gblk, 0, stream>>>(Qh, Kh, Vth, Ohm);
  // x2 = x + attn @ Wo + bo
  gemm_wmma<<<dim3(DM / 64, NROWS / 128), gblk, 0, stream>>>(
      Ohm, Wot, bo, x, x2, nullptr, NROWS, DM, DM, 0);
  // ln2
  ln_kernel<<<NROWS, 256, 0, stream>>>(x2, ln2g, ln2b, xn2h);
  // H1 = relu(xn2 @ W1 + b1)
  gemm_wmma<<<dim3(DFF_ / 64, NROWS / 128), gblk, 0, stream>>>(
      xn2h, W1t, b1, nullptr, nullptr, H1h, NROWS, DFF_, DM, 1);
  // out = x2 + H1 @ W2 + b2
  gemm_wmma<<<dim3(DM / 64, NROWS / 128), gblk, 0, stream>>>(
      H1h, W2t, b2, x2, out, nullptr, NROWS, DM, DFF_, 0);
}
